// MultiHeadAttention_56959856280447
// MI455X (gfx1250) — matrix operand order repaired and checked by reading, not compiled, not run on hardware
//
#include <hip/hip_runtime.h>
#include <hip/hip_bf16.h>

// MHA forward, B=4,S=2048,D=512,H=8,DK=64 on gfx1250 (wave32, WMMA bf16, TDM).
// Roofline: the mandatory 537MB attn write dominates (~23us at 23.3 TB/s);
// K/V are 8MB each in bf16 and stay L2-resident (192MB L2), so attention does
// two passes over K (stats, then recompute+write attn once+fused P@V).
// GEMMs: one-time bf16 conversion (X flat, W transposed) so every WMMA operand
// load is contiguous; A-tiles staged into LDS with the Tensor Data Mover.

typedef __bf16 bf16_t;
typedef __attribute__((ext_vector_type(16))) __bf16 v16bf;
typedef __attribute__((ext_vector_type(8)))  __bf16 v8bf;
typedef __attribute__((ext_vector_type(8)))  float  v8f;
typedef __attribute__((ext_vector_type(4)))  unsigned int u32x4;
typedef __attribute__((ext_vector_type(8)))  int i32x8;
typedef __attribute__((ext_vector_type(4)))  int i32x4;

#define B_   4
#define S_   2048
#define D_   512
#define H_   8
#define DK_  64
#define NROW (B_ * S_)  // 8192

#if defined(__HIP_DEVICE_COMPILE__) && __has_builtin(__builtin_amdgcn_tensor_load_to_lds) && __has_builtin(__builtin_amdgcn_s_wait_tensorcnt)
#define USE_TDM 1
#else
#define USE_TDM 0
#endif

__device__ __forceinline__ v8f wmma_bf16(v16bf a, v16bf b, v8f c) {
  return __builtin_amdgcn_wmma_f32_16x16x32_bf16(
      /*neg_a=*/false, a, /*neg_b=*/false, b,
      /*c_mod=*/(short)0, c, /*reuse_a=*/false, /*reuse_b=*/false);
}

// ---- WMMA register-layout loaders (ISA 7.12.2) -------------------------------
// A (16x32 MxK): lane l holds row m=l&15; K base kb=(l>>4)*16;
// elements [0..15]=K kb..kb+15 (contiguous, matches load_B_kcontig).
__device__ __forceinline__ v16bf load_A_bf(const bf16_t* A, int ld, int k0) {
  const int lane = threadIdx.x & 31;
  const bf16_t* row = A + (lane & 15) * ld + k0;
  const int kb = (lane >> 4) * 16;
  v16bf a;
#pragma unroll
  for (int j = 0; j < 16; ++j) a[j] = row[kb + j];
  return a;
}

// B (32x16 KxN): lane l holds column n=l&15; K base (l>>4)*16 (16 values).
// Source is K-contiguous per column (ld = stride between columns).
__device__ __forceinline__ v16bf load_B_kcontig(const bf16_t* p, int ld) {
  const int lane = threadIdx.x & 31;
  const bf16_t* q = p + (size_t)(lane & 15) * ld + ((lane >> 4) * 16);
  v16bf b;
#pragma unroll
  for (int j = 0; j < 16; ++j) b[j] = q[j];
  return b;
}

// Butterfly reductions across the 16 lanes holding one C-tile row.
__device__ __forceinline__ float rowmax16(float x) {
#pragma unroll
  for (int off = 1; off < 16; off <<= 1) x = fmaxf(x, __shfl_xor(x, off));
  return x;
}
__device__ __forceinline__ float rowsum16(float x) {
#pragma unroll
  for (int off = 1; off < 16; off <<= 1) x += __shfl_xor(x, off);
  return x;
}

// ---- One-time bf16 conversions ----------------------------------------------
// X: [8192,512] fp32 -> bf16 flat (grid.z selects q/k/v), 4 elements/thread.
__global__ __launch_bounds__(256) void mha_cvt_x(
    const float* __restrict__ xq, const float* __restrict__ xk,
    const float* __restrict__ xv, bf16_t* __restrict__ oq,
    bf16_t* __restrict__ ok, bf16_t* __restrict__ ov) {
  const float* src = (blockIdx.z == 0) ? xq : (blockIdx.z == 1) ? xk : xv;
  bf16_t* dst = (blockIdx.z == 0) ? oq : (blockIdx.z == 1) ? ok : ov;
  const size_t i4 = ((size_t)blockIdx.x * blockDim.x + threadIdx.x) * 4;
  const float4 v = *(const float4*)(src + i4);
  dst[i4 + 0] = (bf16_t)v.x;
  dst[i4 + 1] = (bf16_t)v.y;
  dst[i4 + 2] = (bf16_t)v.z;
  dst[i4 + 3] = (bf16_t)v.w;
}

// W: [512,512] fp32 row-major -> Wt[n*512+k] bf16 (transposed, K-contiguous).
__global__ __launch_bounds__(256) void mha_cvt_wt(
    const float* __restrict__ wq, const float* __restrict__ wk,
    const float* __restrict__ wv, const float* __restrict__ wo,
    bf16_t* __restrict__ tq, bf16_t* __restrict__ tk,
    bf16_t* __restrict__ tv, bf16_t* __restrict__ to_) {
  const float* w = (blockIdx.z == 0) ? wq : (blockIdx.z == 1) ? wk
                 : (blockIdx.z == 2) ? wv : wo;
  bf16_t* t = (blockIdx.z == 0) ? tq : (blockIdx.z == 1) ? tk
            : (blockIdx.z == 2) ? tv : to_;
  const int idx = blockIdx.x * 256 + threadIdx.x;  // over 512*512
  const int n = idx >> 9, k = idx & 511;
  t[idx] = (bf16_t)w[(size_t)k * D_ + n];
}

// ---- Staged GEMM: C[64x64] = A[64x512](bf16) @ Wt^T + bias -------------------
// block = 128 (4 waves); wave w owns rows m0+16w; A chunks (64x64, 8KB) staged
// into LDS by the Tensor Data Mover; B tiles contiguous from transposed bf16 W.
// mode 0/1: bf16 head-split [B,H,S,DK]; mode 2: bf16 transposed [B,H,DK,S];
// mode 3: fp32 [M,512] (final output projection).
__global__ __launch_bounds__(128) void mha_gemm64(
    const bf16_t* __restrict__ Abf, const bf16_t* __restrict__ Wt,
    const float* __restrict__ bias, void* __restrict__ dstv, int mode) {
  __shared__ bf16_t atile[64 * 64];
  const int wave = threadIdx.x >> 5;
  const int lane = threadIdx.x & 31;
  const int hi = lane >> 4;
  const int col = lane & 15;
  const int n0 = blockIdx.x * 64;
  const int m0 = blockIdx.y * 64;

  v8f acc[4] = {};
#if USE_TDM
  const unsigned lds_base = (unsigned)(size_t)(void*)atile;
#endif

  for (int c = 0; c < 8; ++c) {  // K = 512 in chunks of 64
#if USE_TDM
    if (wave == 0) {
      // Tensor DMA descriptor (ISA 8.3/8.4): 2D tile 64(k) x 64(rows), bf16.
      const unsigned long long ga =
          (unsigned long long)(const void*)(Abf + (size_t)m0 * D_ + c * 64);
      u32x4 g0;
      g0[0] = 1u;                                   // count=1 (valid D#)
      g0[1] = lds_base;                             // lds_addr
      g0[2] = (unsigned)ga;                         // global_addr[31:0]
      g0[3] = (unsigned)((ga >> 32) & 0x1FFFFFFu)   // global_addr[56:32]
              | (2u << 30);                         // type=2 ("image")
      i32x8 g1;
      g1[0] = (int)(1u << 16);                      // data_size=1 -> 2 bytes
      g1[1] = (int)((unsigned)D_ << 16);            // tensor_dim0[15:0]
      g1[2] = (int)(((unsigned)NROW & 0xFFFFu) << 16);  // tensor_dim1[15:0]
      g1[3] = (int)(((unsigned)NROW >> 16) | (64u << 16)); // dim1 hi | tile_dim0=64
      g1[4] = 64;                                   // tile_dim1=64, tile_dim2=0
      g1[5] = D_;                                   // tensor_dim0_stride=512
      g1[6] = 0;
      g1[7] = 0;
      const i32x4 gz4 = {0, 0, 0, 0};               // groups 2/3 unused (2D)
      const i32x8 gz8 = {0, 0, 0, 0, 0, 0, 0, 0};   // trailing group (unused)
      __builtin_amdgcn_tensor_load_to_lds(g0, g1, gz4, gz4, gz8, 0);
      __builtin_amdgcn_s_wait_tensorcnt((unsigned short)0);
    }
#else
    for (int i = threadIdx.x; i < (64 * 64) / 8; i += 128) {
      const int r = i >> 3, cc = (i & 7) * 8;
      *(v8bf*)(atile + r * 64 + cc) =
          *(const v8bf*)(Abf + (size_t)(m0 + r) * D_ + c * 64 + cc);
    }
#endif
    __syncthreads();
#pragma unroll
    for (int s = 0; s < 2; ++s) {
      const v16bf a = load_A_bf(atile + wave * 16 * 64, 64, s * 32);
#pragma unroll
      for (int t = 0; t < 4; ++t) {
        const v16bf b =
            load_B_kcontig(Wt + (size_t)(n0 + t * 16) * D_ + c * 64 + s * 32, D_);
        acc[t] = wmma_bf16(a, b, acc[t]);
      }
    }
    __syncthreads();
  }

  const int rbase = m0 + wave * 16;
#pragma unroll
  for (int t = 0; t < 4; ++t) {
    const int n = n0 + t * 16 + col;
    const float bv = bias[n];
    const int h = n >> 6, dk = n & 63;
#pragma unroll
    for (int r = 0; r < 8; ++r) {
      const int row = rbase + r + 8 * hi;
      const float v = acc[t][r] + bv;
      if (mode == 3) {
        ((float*)dstv)[(size_t)row * D_ + n] = v;
      } else {
        const int bb = row >> 11, s = row & (S_ - 1);
        bf16_t* dst = (bf16_t*)dstv;
        if (mode == 2)  // V transposed [B,H,DK,S]
          dst[(((size_t)bb * H_ + h) * DK_ + dk) * S_ + s] = (bf16_t)v;
        else            // Q/K head-split [B,H,S,DK]
          dst[(((size_t)bb * H_ + h) * S_ + s) * DK_ + dk] = (bf16_t)v;
      }
    }
  }
}

// ---- Attention: causal, two passes over K, fused P@V -------------------------
// grid = (B*H, S/16/4), block = 128 (4 waves); one wave owns a 16-query tile.
__global__ __launch_bounds__(128) void mha_attention(
    const bf16_t* __restrict__ Qh, const bf16_t* __restrict__ Kh,
    const bf16_t* __restrict__ Vt, float* __restrict__ attn,
    bf16_t* __restrict__ OH) {
  __shared__ bf16_t pbuf[4][16 * 32];  // per-wave P-tile transpose staging

  const int wave = threadIdx.x >> 5;
  const int lane = threadIdx.x & 31;
  const int hi = lane >> 4;
  const int col = lane & 15;

  const int bh = blockIdx.x;
  const int b = bh / H_, h = bh % H_;
  const int qt = blockIdx.y * 4 + wave;
  const int q0 = qt * 16;

  const bf16_t* Qp = Qh + ((size_t)bh * S_ + q0) * DK_;
  const bf16_t* Kp = Kh + (size_t)bh * S_ * DK_;
  const bf16_t* Vp = Vt + (size_t)bh * DK_ * S_;
  const float scale = 0.125f;  // 1/sqrt(64)

  const v16bf aq0 = load_A_bf(Qp, DK_, 0);
  const v16bf aq1 = load_A_bf(Qp, DK_, 32);

  float mrow[8], lrow[8];
#pragma unroll
  for (int r = 0; r < 8; ++r) { mrow[r] = -1e30f; lrow[r] = 0.0f; }

  // pass 1: online softmax statistics (no stores; K tiles stream via L2)
  for (int kt = 0; kt <= qt; ++kt) {
    const int k0 = kt * 16;
    __builtin_prefetch(Kp + (size_t)(k0 + 16) * DK_, 0, 0);
    v8f sc = {};
    sc = wmma_bf16(aq0, load_B_kcontig(Kp + (size_t)k0 * DK_ + 0, DK_), sc);
    sc = wmma_bf16(aq1, load_B_kcontig(Kp + (size_t)k0 * DK_ + 32, DK_), sc);
#pragma unroll
    for (int r = 0; r < 8; ++r) {
      const int q = q0 + r + 8 * hi;
      const int k = k0 + col;
      const float s = (k <= q) ? sc[r] * scale : -1e30f;
      const float tm = rowmax16(s);
      const float nm = fmaxf(mrow[r], tm);
      const float ts = rowsum16(__expf(s - nm));
      lrow[r] = lrow[r] * __expf(mrow[r] - nm) + ts;
      mrow[r] = nm;
    }
  }
  float inv[8];
#pragma unroll
  for (int r = 0; r < 8; ++r) inv[r] = 1.0f / lrow[r];  // diagonal always valid

  // pass 2: recompute scores, write attn exactly once, fused P@V
  float* arow = attn + ((size_t)bh * S_ + q0) * S_;
  v8f oacc[4] = {};
  bf16_t* pb = pbuf[wave];

  for (int kp = 0; kp < S_ / 32; ++kp) {
    const int kbase = kp * 32;
    if (kbase > q0 + 15) {  // fully above diagonal: attn still must be written
#pragma unroll
      for (int r = 0; r < 8; ++r) {
        float* ar = arow + (size_t)(r + 8 * hi) * S_ + kbase + col;
        ar[0] = 0.0f;
        ar[16] = 0.0f;
      }
      continue;
    }
#pragma unroll
    for (int sub = 0; sub < 2; ++sub) {
      const int k0 = kbase + sub * 16;
      v8f sc = {};
      if (k0 <= q0 + 15) {
        sc = wmma_bf16(aq0, load_B_kcontig(Kp + (size_t)k0 * DK_ + 0, DK_), sc);
        sc = wmma_bf16(aq1, load_B_kcontig(Kp + (size_t)k0 * DK_ + 32, DK_), sc);
      }
#pragma unroll
      for (int r = 0; r < 8; ++r) {
        const int q = q0 + r + 8 * hi;
        const int k = k0 + col;
        float p = 0.0f;
        if (k <= q) p = __expf(sc[r] * scale - mrow[r]) * inv[r];
        arow[(size_t)(r + 8 * hi) * S_ + k] = p;
        pb[(r + 8 * hi) * 32 + sub * 16 + col] = (bf16_t)p;  // transpose via LDS
      }
    }
    const v16bf pa = load_A_bf(pb, 32, 0);  // A-layout read of P tile from LDS
#pragma unroll
    for (int t = 0; t < 4; ++t)
      oacc[t] = wmma_bf16(
          pa, load_B_kcontig(Vp + (size_t)t * 16 * S_ + kbase, S_), oacc[t]);
  }

  // merged-head output [B,S,D] in bf16 for the final projection
#pragma unroll
  for (int t = 0; t < 4; ++t) {
#pragma unroll
    for (int r = 0; r < 8; ++r) {
      const int q = q0 + r + 8 * hi;
      const int dk = t * 16 + col;
      OH[((size_t)b * S_ + q) * D_ + h * DK_ + dk] = (bf16_t)oacc[t][r];
    }
  }
}

extern "C" void kernel_launch(void* const* d_in, const int* in_sizes, int n_in,
                              void* d_out, int out_size, void* d_ws,
                              size_t ws_size, hipStream_t stream) {
  const float* query = (const float*)d_in[0];
  const float* key   = (const float*)d_in[1];
  const float* value = (const float*)d_in[2];
  // d_in[3]: causal tril mask, equivalent to k<=q, applied analytically.
  const float* Wq = (const float*)d_in[4];
  const float* bq = (const float*)d_in[5];
  const float* Wk = (const float*)d_in[6];
  const float* bk = (const float*)d_in[7];
  const float* Wv = (const float*)d_in[8];
  const float* bv = (const float*)d_in[9];
  const float* Wo = (const float*)d_in[10];
  const float* bo = (const float*)d_in[11];

  float* out  = (float*)d_out;               // [B,S,D]
  float* attn = out + (size_t)B_ * S_ * D_;  // [B,H,S,S]

  const size_t XN = (size_t)NROW * D_;  // 4,194,304 elements
  bf16_t* Xqbf = (bf16_t*)d_ws;         // [8192,512]; reused later as OH
  bf16_t* Xkbf = Xqbf + XN;
  bf16_t* Xvbf = Xkbf + XN;
  bf16_t* Qh   = Xvbf + XN;             // [B,H,S,DK]
  bf16_t* Kh   = Qh + XN;               // [B,H,S,DK]
  bf16_t* Vt   = Kh + XN;               // [B,H,DK,S]
  bf16_t* Wqt  = Vt + XN;               // [512,512] transposed
  bf16_t* Wkt  = Wqt + (size_t)D_ * D_;
  bf16_t* Wvt  = Wkt + (size_t)D_ * D_;
  bf16_t* Wot  = Wvt + (size_t)D_ * D_;
  bf16_t* OH   = Xqbf;                  // alias: Xqbf dead after Q projection

  mha_cvt_x<<<dim3(XN / (256 * 4), 1, 3), 256, 0, stream>>>(
      query, key, value, Xqbf, Xkbf, Xvbf);
  mha_cvt_wt<<<dim3((D_ * D_) / 256, 1, 4), 256, 0, stream>>>(
      Wq, Wk, Wv, Wo, Wqt, Wkt, Wvt, Wot);

  dim3 gg(D_ / 64, NROW / 64);
  mha_gemm64<<<gg, 128, 0, stream>>>(Xqbf, Wqt, bq, Qh, 0);
  mha_gemm64<<<gg, 128, 0, stream>>>(Xkbf, Wkt, bk, Kh, 1);
  mha_gemm64<<<gg, 128, 0, stream>>>(Xvbf, Wvt, bv, Vt, 2);

  mha_attention<<<dim3(B_ * H_, (S_ / 16) / 4), 128, 0, stream>>>(
      Qh, Kh, Vt, attn, OH);

  mha_gemm64<<<gg, 128, 0, stream>>>(OH, Wot, bo, out, 3);
}
